// MultiheadAttention_19464791786091
// MI455X (gfx1250) — matrix operand order repaired and checked by reading, not compiled, not run on hardware
//
#include <hip/hip_runtime.h>
#include <hip/hip_bf16.h>
#include <math.h>

// ---------------------------------------------------------------------------
// Problem constants
// ---------------------------------------------------------------------------
#define NB   2
#define LSEQ 2048
#define DIM  1024
#define NH   16
#define NKV  4
#define HD   64
#define CAPV 50.0f

typedef _Float16 half8  __attribute__((ext_vector_type(8)));
typedef _Float16 half16 __attribute__((ext_vector_type(16)));
typedef float    float8 __attribute__((ext_vector_type(8)));

__device__ __forceinline__ float fast_tanh(float x) {
#if __has_builtin(__builtin_amdgcn_tanhf)
  return __builtin_amdgcn_tanhf(x);   // V_TANH_F32 (CDNA5 trans op)
#else
  return tanhf(x);
#endif
}

// ---------------------------------------------------------------------------
// WMMA wrapper: D = A(16x32 f16) * B(32x16 f16) + C(16x16 f32)
// ---------------------------------------------------------------------------
__device__ __forceinline__ float8 wmma_f32_16x16x32(half16 a, half16 b, float8 c) {
  return __builtin_amdgcn_wmma_f32_16x16x32_f16(false, a, false, b, (short)0, c, false, false);
}

// A fragment (16x32, f16, row-major source with leading dim ld):
// lane m = lane&15 is row; h = lane>>4; elems 0..7 = K k0+8h.., elems 8..15 = K k0+16+8h..
__device__ __forceinline__ half16 load_a_frag(const _Float16* __restrict__ base, int ld,
                                              int row0, int k0, int lane) {
  int m = lane & 15, h = lane >> 4;
  const _Float16* p = base + (size_t)(row0 + m) * ld + k0 + 8 * h;
  half8 lo = *(const half8*)p;
  half8 hi = *(const half8*)(p + 16);
  half16 a;
#pragma unroll
  for (int i = 0; i < 8; ++i) { a[i] = lo[i]; a[i + 8] = hi[i]; }
  return a;
}

// B fragment (32x16, f16). Source layout: [n][k] (k contiguous, leading dim ld).
// lane n = lane&15 is column; h = lane>>4; elems 0..7 = K k0+8h.., elems 8..15 = K k0+16+8h..
__device__ __forceinline__ half16 load_b_frag(const _Float16* __restrict__ base, int ld,
                                              int col0, int k0, int lane) {
  int n = lane & 15, h = lane >> 4;
  const _Float16* p = base + (size_t)(col0 + n) * ld + k0 + 8 * h;
  half8 lo = *(const half8*)p;
  half8 hi = *(const half8*)(p + 16);
  half16 b;
#pragma unroll
  for (int i = 0; i < 8; ++i) { b[i] = lo[i]; b[i + 8] = hi[i]; }
  return b;
}

// ---------------------------------------------------------------------------
// Stage 0: fp32 -> f16 conversions
// ---------------------------------------------------------------------------
__global__ void convert_x_kernel(const float* __restrict__ x, _Float16* __restrict__ xh, int n) {
  int i = blockIdx.x * 256 + threadIdx.x;
  if (i < n) xh[i] = (_Float16)x[i];
}

// W: [K][C] fp32 -> Wt: [C][K] f16
__global__ void transpose_w_kernel(const float* __restrict__ W, _Float16* __restrict__ Wt,
                                   int K, int C) {
  int i = blockIdx.x * 256 + threadIdx.x;
  if (i < K * C) {
    int k = i / C, c = i % C;
    Wt[(size_t)c * K + k] = (_Float16)W[i];
  }
}

// ---------------------------------------------------------------------------
// Stage 1: fused QKV projection + bias + RMSNorm + RoPE.
// grid = (16 row-tiles of 256, 24 head-slots: 16 Q, 4 K, 4 V), block = 256 (8 waves).
// Each wave: 32 rows x one head's 64 columns. K loop is a 2-deep ping-pong
// (buffers A/B hold even/odd 32-chunks) so no register copies are needed and
// each chunk's loads are in flight across the other chunk's 8 WMMAs.
// Q/K -> row-major [b][h][l][64], V -> transposed [b][kv][d][L].
// ---------------------------------------------------------------------------
__global__ __launch_bounds__(256) void qkv_kernel(
    const _Float16* __restrict__ xh,
    const _Float16* __restrict__ Wqt, const _Float16* __restrict__ Wkt,
    const _Float16* __restrict__ Wvt,
    const float* __restrict__ bq, const float* __restrict__ bk, const float* __restrict__ bv,
    const float* __restrict__ qg, const float* __restrict__ kg,
    const float* __restrict__ rcos, const float* __restrict__ rsin,
    _Float16* __restrict__ Qh, _Float16* __restrict__ Kh, _Float16* __restrict__ Vt) {
  const int w = threadIdx.x >> 5, lane = threadIdx.x & 31;
  const int n = lane & 15, hc = lane >> 4;
  const int row0 = blockIdx.x * 256 + w * 32;
  const int slot = blockIdx.y;

  int type, head;
  const _Float16* Wt;
  const float* bias;
  const float* gamma = nullptr;
  if (slot < NH)            { type = 0; head = slot;       Wt = Wqt + (size_t)head * HD * DIM; bias = bq + head * HD; gamma = qg + head * HD; }
  else if (slot < NH + NKV) { type = 1; head = slot - NH;  Wt = Wkt + (size_t)head * HD * DIM; bias = bk + head * HD; gamma = kg + head * HD; }
  else                      { type = 2; head = slot - NH - NKV; Wt = Wvt + (size_t)head * HD * DIM; bias = bv + head * HD; }

  float8 acc[2][4] = {};
  half16 aA[2], bA[4], aB[2], bB[4];
  aA[0] = load_a_frag(xh, DIM, row0, 0, lane);
  aA[1] = load_a_frag(xh, DIM, row0 + 16, 0, lane);
#pragma unroll
  for (int nb = 0; nb < 4; ++nb) bA[nb] = load_b_frag(Wt, DIM, nb * 16, 0, lane);

  int kk = 0;
  for (; kk + 64 < DIM; kk += 64) {
    // odd-chunk loads (kk+32) in flight across even-chunk WMMAs
    aB[0] = load_a_frag(xh, DIM, row0, kk + 32, lane);
    aB[1] = load_a_frag(xh, DIM, row0 + 16, kk + 32, lane);
#pragma unroll
    for (int nb = 0; nb < 4; ++nb) bB[nb] = load_b_frag(Wt, DIM, nb * 16, kk + 32, lane);
#pragma unroll
    for (int nb = 0; nb < 4; ++nb) {
      acc[0][nb] = wmma_f32_16x16x32(aA[0], bA[nb], acc[0][nb]);
      acc[1][nb] = wmma_f32_16x16x32(aA[1], bA[nb], acc[1][nb]);
    }
    // next even-chunk loads (kk+64) in flight across odd-chunk WMMAs
    aA[0] = load_a_frag(xh, DIM, row0, kk + 64, lane);
    aA[1] = load_a_frag(xh, DIM, row0 + 16, kk + 64, lane);
#pragma unroll
    for (int nb = 0; nb < 4; ++nb) bA[nb] = load_b_frag(Wt, DIM, nb * 16, kk + 64, lane);
#pragma unroll
    for (int nb = 0; nb < 4; ++nb) {
      acc[0][nb] = wmma_f32_16x16x32(aB[0], bB[nb], acc[0][nb]);
      acc[1][nb] = wmma_f32_16x16x32(aB[1], bB[nb], acc[1][nb]);
    }
  }
  // tail: chunks kk (already in A) and kk+32
  aB[0] = load_a_frag(xh, DIM, row0, kk + 32, lane);
  aB[1] = load_a_frag(xh, DIM, row0 + 16, kk + 32, lane);
#pragma unroll
  for (int nb = 0; nb < 4; ++nb) bB[nb] = load_b_frag(Wt, DIM, nb * 16, kk + 32, lane);
#pragma unroll
  for (int nb = 0; nb < 4; ++nb) {
    acc[0][nb] = wmma_f32_16x16x32(aA[0], bA[nb], acc[0][nb]);
    acc[1][nb] = wmma_f32_16x16x32(aA[1], bA[nb], acc[1][nb]);
  }
#pragma unroll
  for (int nb = 0; nb < 4; ++nb) {
    acc[0][nb] = wmma_f32_16x16x32(aB[0], bB[nb], acc[0][nb]);
    acc[1][nb] = wmma_f32_16x16x32(aB[1], bB[nb], acc[1][nb]);
  }

  // bias
#pragma unroll
  for (int nb = 0; nb < 4; ++nb) {
    float bvv = bias[nb * 16 + n];
#pragma unroll
    for (int r = 0; r < 2; ++r)
#pragma unroll
      for (int i = 0; i < 8; ++i) acc[r][nb][i] += bvv;
  }

  if (type != 2) {
#pragma unroll
    for (int r = 0; r < 2; ++r) {
      // RMSNorm over HD=64: row spread over 16 lanes -> xor-reduce 1,2,4,8.
      float inv[8];
#pragma unroll
      for (int i = 0; i < 8; ++i) {
        float ss = acc[r][0][i] * acc[r][0][i] + acc[r][1][i] * acc[r][1][i] +
                   acc[r][2][i] * acc[r][2][i] + acc[r][3][i] * acc[r][3][i];
        ss += __shfl_xor(ss, 1); ss += __shfl_xor(ss, 2);
        ss += __shfl_xor(ss, 4); ss += __shfl_xor(ss, 8);
        inv[i] = rsqrtf(ss * (1.0f / 64.0f) + 1e-6f);
      }
#pragma unroll
      for (int nb = 0; nb < 4; ++nb) {
        float g = gamma[nb * 16 + n];
#pragma unroll
        for (int i = 0; i < 8; ++i) acc[r][nb][i] *= inv[i] * g;
      }
      // RoPE: column d = nb*16+n pairs with d+32 = (nb+2)*16+n — same lane.
#pragma unroll
      for (int i = 0; i < 8; ++i) {
        int pos = (row0 + r * 16 + i + 8 * hc) & (LSEQ - 1);
#pragma unroll
        for (int nb = 0; nb < 2; ++nb) {
          int d = nb * 16 + n;
          float c1 = rcos[pos * HD + d],      s1 = rsin[pos * HD + d];
          float c2 = rcos[pos * HD + d + 32], s2 = rsin[pos * HD + d + 32];
          float lo = acc[r][nb][i], hi = acc[r][nb + 2][i];
          acc[r][nb][i]     = lo * c1 - hi * s1;
          acc[r][nb + 2][i] = hi * c2 + lo * s2;
        }
      }
    }
  }

  // store
#pragma unroll
  for (int r = 0; r < 2; ++r) {
#pragma unroll
    for (int i = 0; i < 8; ++i) {
      int rr = row0 + r * 16 + i + 8 * hc;
      int bb = rr >> 11, l = rr & (LSEQ - 1);
      if (type == 0) {
        _Float16* dst = Qh + (((size_t)bb * NH + head) * LSEQ + l) * HD;
#pragma unroll
        for (int nb = 0; nb < 4; ++nb) dst[nb * 16 + n] = (_Float16)acc[r][nb][i];
      } else if (type == 1) {
        _Float16* dst = Kh + (((size_t)bb * NKV + head) * LSEQ + l) * HD;
#pragma unroll
        for (int nb = 0; nb < 4; ++nb) dst[nb * 16 + n] = (_Float16)acc[r][nb][i];
      } else {
        _Float16* dst = Vt + (((size_t)bb * NKV + head) * HD) * LSEQ + l;
#pragma unroll
        for (int nb = 0; nb < 4; ++nb) dst[(size_t)(nb * 16 + n) * LSEQ] = (_Float16)acc[r][nb][i];
      }
    }
  }
}

// ---------------------------------------------------------------------------
// Stage 2: causal attention with tanh soft-cap.
// grid = (LSEQ/128, NB*NH), block = 256 (8 waves). Wave = 16 queries x HD=64.
// Soft-cap bounds logits <= CAP, so softmax uses fixed max CAP: p = exp(s-CAP),
// no online rescaling. Key loop steps 64 keys as two ping-pong 32-key tiles
// (K-fragment sets A/B, no register copies); V loads and next-tile K loads are
// issued ahead of the tanh/exp VALU block so transcendentals overlap memory.
// ---------------------------------------------------------------------------
__global__ __launch_bounds__(256) void attn_kernel(
    const _Float16* __restrict__ Qh, const _Float16* __restrict__ Kh,
    const _Float16* __restrict__ Vt, _Float16* __restrict__ Oh,
    const int* __restrict__ is_causal) {
  __shared__ _Float16 pshm[8][16][40];   // per-wave 16x32 P tile, padded rows

  const int w = threadIdx.x >> 5, lane = threadIdx.x & 31;
  const int n = lane & 15, hq = lane >> 4;
  const int bh = blockIdx.y;
  const int b = bh >> 4, hh = bh & 15;
  const int kvh = hh >> 2;                       // GQA: 4 Q heads per KV head
  const int qb = blockIdx.x * 128 + w * 16;
  const int causal = *is_causal;

  const _Float16* Qbase = Qh + (((size_t)b * NH + hh) * LSEQ) * HD;
  const _Float16* Kbase = Kh + (((size_t)b * NKV + kvh) * LSEQ) * HD;
  const _Float16* Vbase = Vt + (((size_t)b * NKV + kvh) * HD) * LSEQ;

  half16 aq0 = load_a_frag(Qbase, HD, qb, 0, lane);
  half16 aq1 = load_a_frag(Qbase, HD, qb, 32, lane);

  float8 acc[4] = {};
  float rsum[8] = {};
  const float scale = 0.125f;            // HD^-0.5
  const float invcap = 1.0f / CAPV;
  const int kend = causal ? (qb + 15) : (LSEQ - 1);

  // soft-cap + fixed-max softmax numerator for one 32-key tile at kc0,
  // stage P through LDS (C -> A layout), accumulate P*V.
  auto process_tile = [&](int kc0, const float8 (&s)[2], const half16 (&bvf)[4]) {
#pragma unroll
    for (int sb = 0; sb < 2; ++sb) {
#pragma unroll
      for (int i = 0; i < 8; ++i) {
        int q = qb + i + 8 * hq;
        int key = kc0 + sb * 16 + n;
        int qlim = causal ? q : (LSEQ - 1);
        float sv = s[sb][i] * scale;
        float t = CAPV * fast_tanh(sv * invcap);
        float p = (key <= qlim) ? __expf(t - CAPV) : 0.0f;
        rsum[i] += p;
        pshm[w][i + 8 * hq][sb * 16 + n] = (_Float16)p;
      }
    }
    asm volatile("s_wait_dscnt 0" ::: "memory");
    half16 pa;
    {
      int m = lane & 15, h = lane >> 4;
      const _Float16* pr = &pshm[w][m][0];
#pragma unroll
      for (int e = 0; e < 8; ++e) { pa[e] = pr[8 * h + e]; pa[e + 8] = pr[16 + 8 * h + e]; }
    }
#pragma unroll
    for (int nb = 0; nb < 4; ++nb) acc[nb] = wmma_f32_16x16x32(pa, bvf[nb], acc[nb]);
  };

  // preload K fragments for tile 0 into set A: bk[sb][chunk]
  half16 bkA[2][2], bkB[2][2];
#pragma unroll
  for (int sb = 0; sb < 2; ++sb)
#pragma unroll
    for (int c = 0; c < 2; ++c)
      bkA[sb][c] = load_b_frag(Kbase, HD, sb * 16, c * 32, lane);

  for (int k0 = 0; k0 <= kend; k0 += 64) {
    const int k1 = k0 + 32;
    const bool hasT1 = (k1 <= kend);     // wave-uniform

    // tile 0: V frags + (uniform) prefetch of tile-1 K frags into set B
    half16 bv0[4];
#pragma unroll
    for (int nb = 0; nb < 4; ++nb) bv0[nb] = load_b_frag(Vbase, LSEQ, nb * 16, k0, lane);
    if (hasT1) {
#pragma unroll
      for (int sb = 0; sb < 2; ++sb)
#pragma unroll
        for (int c = 0; c < 2; ++c)
          bkB[sb][c] = load_b_frag(Kbase, HD, k1 + sb * 16, c * 32, lane);
    }
    float8 s0[2];
#pragma unroll
    for (int sb = 0; sb < 2; ++sb) {
      float8 z = {};
      z = wmma_f32_16x16x32(aq0, bkA[sb][0], z);
      z = wmma_f32_16x16x32(aq1, bkA[sb][1], z);
      s0[sb] = z;
    }
    // prefetch next iteration's tile-0 K frags into set A
    if (k0 + 64 <= kend) {
#pragma unroll
      for (int sb = 0; sb < 2; ++sb)
#pragma unroll
        for (int c = 0; c < 2; ++c)
          bkA[sb][c] = load_b_frag(Kbase, HD, k0 + 64 + sb * 16, c * 32, lane);
    }
    process_tile(k0, s0, bv0);

    if (hasT1) {
      half16 bv1[4];
#pragma unroll
      for (int nb = 0; nb < 4; ++nb) bv1[nb] = load_b_frag(Vbase, LSEQ, nb * 16, k1, lane);
      float8 s1[2];
#pragma unroll
      for (int sb = 0; sb < 2; ++sb) {
        float8 z = {};
        z = wmma_f32_16x16x32(aq0, bkB[sb][0], z);
        z = wmma_f32_16x16x32(aq1, bkB[sb][1], z);
        s1[sb] = z;
      }
      process_tile(k1, s1, bv1);
    }
  }

  // finalize: full row sums (xor-reduce across the 16-lane group), normalize
#pragma unroll
  for (int i = 0; i < 8; ++i) {
    float s = rsum[i];
    s += __shfl_xor(s, 1); s += __shfl_xor(s, 2);
    s += __shfl_xor(s, 4); s += __shfl_xor(s, 8);
    rsum[i] = 1.0f / s;
  }
#pragma unroll
  for (int i = 0; i < 8; ++i) {
    int q = qb + i + 8 * hq;
    _Float16* dst = Oh + ((size_t)b * LSEQ + q) * DIM + hh * HD;
#pragma unroll
    for (int nb = 0; nb < 4; ++nb) dst[nb * 16 + n] = (_Float16)(acc[nb][i] * rsum[i]);
  }
}

// ---------------------------------------------------------------------------
// Stage 3: output projection: out(4096x1024 f32) = Oh @ Wo + bo
// grid = (4096/256, 1024/64), block = 256 (8 waves x 32 rows x 64 cols).
// Same ping-pong pipeline as qkv.
// ---------------------------------------------------------------------------
__global__ __launch_bounds__(256) void proj_kernel(
    const _Float16* __restrict__ Oh, const _Float16* __restrict__ Wot,
    const float* __restrict__ bo, float* __restrict__ out) {
  const int w = threadIdx.x >> 5, lane = threadIdx.x & 31;
  const int n = lane & 15, h = lane >> 4;
  const int row0 = blockIdx.x * 256 + w * 32;
  const int col0 = blockIdx.y * 64;

  float8 acc[2][4] = {};
  half16 aA[2], bA[4], aB[2], bB[4];
  aA[0] = load_a_frag(Oh, DIM, row0, 0, lane);
  aA[1] = load_a_frag(Oh, DIM, row0 + 16, 0, lane);
#pragma unroll
  for (int nb = 0; nb < 4; ++nb) bA[nb] = load_b_frag(Wot, DIM, col0 + nb * 16, 0, lane);

  int kk = 0;
  for (; kk + 64 < DIM; kk += 64) {
    aB[0] = load_a_frag(Oh, DIM, row0, kk + 32, lane);
    aB[1] = load_a_frag(Oh, DIM, row0 + 16, kk + 32, lane);
#pragma unroll
    for (int nb = 0; nb < 4; ++nb) bB[nb] = load_b_frag(Wot, DIM, col0 + nb * 16, kk + 32, lane);
#pragma unroll
    for (int nb = 0; nb < 4; ++nb) {
      acc[0][nb] = wmma_f32_16x16x32(aA[0], bA[nb], acc[0][nb]);
      acc[1][nb] = wmma_f32_16x16x32(aA[1], bA[nb], acc[1][nb]);
    }
    aA[0] = load_a_frag(Oh, DIM, row0, kk + 64, lane);
    aA[1] = load_a_frag(Oh, DIM, row0 + 16, kk + 64, lane);
#pragma unroll
    for (int nb = 0; nb < 4; ++nb) bA[nb] = load_b_frag(Wot, DIM, col0 + nb * 16, kk + 64, lane);
#pragma unroll
    for (int nb = 0; nb < 4; ++nb) {
      acc[0][nb] = wmma_f32_16x16x32(aB[0], bB[nb], acc[0][nb]);
      acc[1][nb] = wmma_f32_16x16x32(aB[1], bB[nb], acc[1][nb]);
    }
  }
  aB[0] = load_a_frag(Oh, DIM, row0, kk + 32, lane);
  aB[1] = load_a_frag(Oh, DIM, row0 + 16, kk + 32, lane);
#pragma unroll
  for (int nb = 0; nb < 4; ++nb) bB[nb] = load_b_frag(Wot, DIM, col0 + nb * 16, kk + 32, lane);
#pragma unroll
  for (int nb = 0; nb < 4; ++nb) {
    acc[0][nb] = wmma_f32_16x16x32(aA[0], bA[nb], acc[0][nb]);
    acc[1][nb] = wmma_f32_16x16x32(aA[1], bA[nb], acc[1][nb]);
  }
#pragma unroll
  for (int nb = 0; nb < 4; ++nb) {
    acc[0][nb] = wmma_f32_16x16x32(aB[0], bB[nb], acc[0][nb]);
    acc[1][nb] = wmma_f32_16x16x32(aB[1], bB[nb], acc[1][nb]);
  }

#pragma unroll
  for (int r = 0; r < 2; ++r) {
#pragma unroll
    for (int i = 0; i < 8; ++i) {
      int rr = row0 + r * 16 + i + 8 * h;
      float* dst = out + (size_t)rr * DIM + col0;
#pragma unroll
      for (int nb = 0; nb < 4; ++nb) dst[nb * 16 + n] = acc[r][nb][i] + bo[col0 + nb * 16 + n];
    }
  }
}

// ---------------------------------------------------------------------------
// Host launcher
// ---------------------------------------------------------------------------
extern "C" void kernel_launch(void* const* d_in, const int* in_sizes, int n_in,
                              void* d_out, int out_size, void* d_ws, size_t ws_size,
                              hipStream_t stream) {
  const float* x   = (const float*)d_in[0];
  const float* Wq  = (const float*)d_in[1];
  const float* bq  = (const float*)d_in[2];
  const float* Wk  = (const float*)d_in[3];
  const float* bk  = (const float*)d_in[4];
  const float* Wv  = (const float*)d_in[5];
  const float* bv  = (const float*)d_in[6];
  const float* Wo  = (const float*)d_in[7];
  const float* bo  = (const float*)d_in[8];
  const float* qg  = (const float*)d_in[9];
  const float* kg  = (const float*)d_in[10];
  const float* rc  = (const float*)d_in[11];
  const float* rs  = (const float*)d_in[12];
  const int* causl = (const int*)d_in[13];
  float* out = (float*)d_out;

  // workspace layout (f16 elements), padded; Vt placed right before Oh so any
  // tail overreads of K/V tiles land on finite f16 data.
  const size_t PAD = 128;
  _Float16* ws = (_Float16*)d_ws;
  size_t off = 0;
  _Float16* xh  = ws + off; off += (size_t)NB * LSEQ * DIM + PAD;
  _Float16* Wqt = ws + off; off += (size_t)DIM * DIM + PAD;
  _Float16* Wkt = ws + off; off += (size_t)(NKV * HD) * DIM + PAD;
  _Float16* Wvt = ws + off; off += (size_t)(NKV * HD) * DIM + PAD;
  _Float16* Wot = ws + off; off += (size_t)DIM * DIM + PAD;
  _Float16* Qh  = ws + off; off += (size_t)NB * NH * LSEQ * HD + PAD;
  _Float16* Kh  = ws + off; off += (size_t)NB * NKV * LSEQ * HD + PAD;
  _Float16* Vt  = ws + off; off += (size_t)NB * NKV * HD * LSEQ + PAD;
  _Float16* Oh  = ws + off; off += (size_t)NB * LSEQ * DIM + PAD;

  // Stage 0: conversions / transposes
  {
    int nx = NB * LSEQ * DIM;
    convert_x_kernel<<<(nx + 255) / 256, 256, 0, stream>>>(x, xh, nx);
    transpose_w_kernel<<<(DIM * DIM + 255) / 256, 256, 0, stream>>>(Wq, Wqt, DIM, DIM);
    transpose_w_kernel<<<(DIM * NKV * HD + 255) / 256, 256, 0, stream>>>(Wk, Wkt, DIM, NKV * HD);
    transpose_w_kernel<<<(DIM * NKV * HD + 255) / 256, 256, 0, stream>>>(Wv, Wvt, DIM, NKV * HD);
    transpose_w_kernel<<<(DIM * DIM + 255) / 256, 256, 0, stream>>>(Wo, Wot, DIM, DIM);
  }

  // Stage 1: QKV + RMSNorm + RoPE
  {
    dim3 grid((NB * LSEQ) / 256, NH + 2 * NKV);   // (16, 24)
    qkv_kernel<<<grid, 256, 0, stream>>>(xh, Wqt, Wkt, Wvt, bq, bk, bv,
                                         qg, kg, rc, rs, Qh, Kh, Vt);
  }

  // Stage 2: attention
  {
    dim3 grid(LSEQ / 128, NB * NH);               // (16, 32)
    attn_kernel<<<grid, 256, 0, stream>>>(Qh, Kh, Vt, Oh, causl);
  }

  // Stage 3: output projection
  {
    dim3 grid((NB * LSEQ) / 256, DIM / 64);       // (16, 16)
    proj_kernel<<<grid, 256, 0, stream>>>(Oh, Wot, bo, out);
  }
}
